// HydraBlock_52750788329955
// MI455X (gfx1250) — matrix operand order repaired and checked by reading, not compiled, not run on hardware
//
#include <hip/hip_runtime.h>

#define D_    1024
#define H_    16
#define T_    2048
#define B_    2
#define DH_   64
#define LAT_  32
#define E_    4
#define EDIM_ 2048
#define NTOK  (B_ * T_)

typedef __attribute__((ext_vector_type(16))) __bf16 v16bf;
typedef __attribute__((ext_vector_type(8)))  float  v8f;
typedef unsigned int  v4u  __attribute__((ext_vector_type(4)));
typedef int           v8i_t __attribute__((ext_vector_type(8)));
typedef int           v4i_t __attribute__((ext_vector_type(4)));

union BF16Frag { v16bf v; unsigned int u[8]; };

__device__ __forceinline__ unsigned short f2bf(float f) {
  unsigned int u = __builtin_bit_cast(unsigned int, f);
  u += 0x7FFFu + ((u >> 16) & 1u);            // round-to-nearest-even
  return (unsigned short)(u >> 16);
}

#if defined(__has_builtin)
#if __has_builtin(__builtin_amdgcn_tensor_load_to_lds) && \
    __has_builtin(__builtin_amdgcn_s_wait_tensorcnt)
#define USE_TDM 1
#endif
#endif

#ifdef USE_TDM
// 2D bf16 tile load, global -> LDS, via Tensor Data Mover.
// Tile is tileX elements wide (dim0), tileY rows (dim1), row stride strideX
// elements. Tile must be fully in-bounds (tensor dims set == tile dims).
// D# layout per CDNA5 ISA 8.3-8.6: group0 {count=1, lds_addr, global_addr,
// type=2}; group1 {data_size=2B, tensor_dim0/1, tile_dim0/1, dim0_stride}.
// This toolchain exposes the 6-arg builtin:
//   (uint32x4 g0, int32x8 g1, int32x4 g2, int32x4 g3, int32x8, i32 cpol)
__device__ __forceinline__ void tdm_load_2d_bf16(unsigned lds_off,
                                                 const void* gptr,
                                                 unsigned tileX, unsigned tileY,
                                                 unsigned strideX)
{
  unsigned long long ga = (unsigned long long)(size_t)gptr;
  v4u g0;
  g0.x = 1u;                                         // count=1, user mode
  g0.y = lds_off;                                    // LDS byte address
  g0.z = (unsigned)(ga & 0xFFFFFFFFu);               // global_addr[31:0]
  g0.w = (unsigned)((ga >> 32) & 0x01FFFFFFu)        // global_addr[56:32]
       | (2u << 30);                                 // type=2 ("image")
  v8i_t g1;
  g1[0] = (int)(1u << 16);                           // data_size=1 -> 2 bytes
  g1[1] = (int)((tileX & 0xFFFFu) << 16);            // tensor_dim0[15:0] @63:48
  g1[2] = (int)(((tileX >> 16) & 0xFFFFu)            // tensor_dim0[31:16]
       |        ((tileY & 0xFFFFu) << 16));          // tensor_dim1[15:0]
  g1[3] = (int)(((tileY >> 16) & 0xFFFFu)            // tensor_dim1[31:16]
       |        ((tileX & 0xFFFFu) << 16));          // tile_dim0
  g1[4] = (int)(tileY & 0xFFFFu);                    // tile_dim1 (tile_dim2=0)
  g1[5] = (int)strideX;                              // tensor_dim0_stride[31:0]
  g1[6] = 0;                                         // stride hi / dim1_stride
  g1[7] = 0;
  v4i_t g2 = {0, 0, 0, 0};
  v4i_t g3 = {0, 0, 0, 0};
  v8i_t g4 = {0, 0, 0, 0, 0, 0, 0, 0};
  __builtin_amdgcn_tensor_load_to_lds(g0, g1, g2, g3, g4, 0);
}
#endif

// ---------------------------------------------------------------------------
// fp32 -> packed bf16 conversion (one-time per tensor)
// ---------------------------------------------------------------------------
__global__ __launch_bounds__(256) void cvt_f32_bf16(
    const float* __restrict__ in, unsigned short* __restrict__ out,
    long long n4)
{
  long long i = (long long)blockIdx.x * 256 + threadIdx.x;
  const long long stride = (long long)gridDim.x * 256;
  for (; i < n4; i += stride) {
    float4 v = *(const float4*)(in + i * 4);
    uint2 p;
    p.x = (unsigned)f2bf(v.x) | ((unsigned)f2bf(v.y) << 16);
    p.y = (unsigned)f2bf(v.z) | ((unsigned)f2bf(v.w) << 16);
    *(uint2*)(out + i * 4) = p;
  }
}

// ---------------------------------------------------------------------------
// Generic GEMM: A[M,K] (bf16) @ B[K,N] (bf16) -> fp32 accum.
// mode 0: val+bias; mode 1: exact GELU(val+bias);
// mode 2: Cf = (beta ? Cf : 0) + rowScale[m*rsStride]*(val+bias)
// Cf (fp32) and Cbf (bf16) destinations each optional.
// A tile (128x32, always in-bounds) streamed by TDM when available.
// ---------------------------------------------------------------------------
__global__ __launch_bounds__(256) void gemm_wmma_bf16(
    const unsigned short* __restrict__ A, const unsigned short* __restrict__ B,
    float* __restrict__ Cf, unsigned short* __restrict__ Cbf,
    int M, int N, int K,
    const float* __restrict__ bias, int mode,
    const float* __restrict__ rowScale, int rsStride, int beta)
{
  __shared__ __align__(16) unsigned short Al[128][32];   // [m][k]
  __shared__ __align__(16) unsigned short Bl[128][32];   // [n][k] (transposed)

  const int tid  = threadIdx.x;
  const int lane = tid & 31, wave = tid >> 5;
  const int half = lane >> 4, l16 = lane & 15;
  const int m0 = blockIdx.y * 128;
  const int n0 = blockIdx.x * 128;

  const v8f vzero = {0.f,0.f,0.f,0.f,0.f,0.f,0.f,0.f};
  v8f acc[8];
#pragma unroll
  for (int s = 0; s < 8; ++s) acc[s] = vzero;

  for (int k0 = 0; k0 < K; k0 += 32) {
#ifdef USE_TDM
    if (wave == 0) {
      tdm_load_2d_bf16((unsigned)(size_t)&Al[0][0],
                       A + (size_t)m0 * K + k0, 32u, 128u, (unsigned)K);
    }
#else
    // cooperative A copy: 512 uint4 slots (8 bf16 each)
#pragma unroll
    for (int i = 0; i < 2; ++i) {
      int slot = tid + i * 256;
      int r = slot >> 2, c = (slot & 3) << 3;
      *(uint4*)&Al[r][c] =
          *(const uint4*)(A + (size_t)(m0 + r) * K + (k0 + c));
    }
#endif
    // stage B tile 32x128 transposed -> Bl[n][k]; 2048 u32 slots (2 bf16)
#pragma unroll
    for (int i = 0; i < 8; ++i) {
      int slot = tid + i * 256;
      int kr = slot >> 6, nc = (slot & 63) << 1;
      unsigned v = 0;
      if (n0 + nc < N)
        v = *(const unsigned int*)(B + (size_t)(k0 + kr) * N + (n0 + nc));
      Bl[nc + 0][kr] = (unsigned short)(v & 0xFFFFu);
      Bl[nc + 1][kr] = (unsigned short)(v >> 16);
    }
#ifdef USE_TDM
    if (wave == 0) __builtin_amdgcn_s_wait_tensorcnt((short)0);
#endif
    __syncthreads();

    BF16Frag af;
    const int mr = wave * 16 + l16;
#pragma unroll
    for (int p = 0; p < 8; ++p) {
      int kk = 2 * p + 8 * half + (p >= 4 ? 8 : 0);
      af.u[p] = *(const unsigned int*)&Al[mr][kk];
    }
#pragma unroll
    for (int s = 0; s < 8; ++s) {
      BF16Frag bfr;
      const int nr = s * 16 + l16;
#pragma unroll
      for (int p = 0; p < 8; ++p) {
        int kk = 2 * p + 8 * half + (p >= 4 ? 8 : 0);
        bfr.u[p] = *(const unsigned int*)&Bl[nr][kk];
      }
      acc[s] = __builtin_amdgcn_wmma_f32_16x16x32_bf16(
          false, af.v, false, bfr.v, (short)0, acc[s], false, false);
    }
    __syncthreads();
  }

  // epilogue
#pragma unroll
  for (int s = 0; s < 8; ++s) {
    int n = n0 + s * 16 + l16;
    if (n >= N) continue;
    float bv = bias ? bias[n] : 0.f;
#pragma unroll
    for (int r = 0; r < 8; ++r) {
      int m = m0 + wave * 16 + r + 8 * half;
      float val = acc[s][r] + bv;
      if (mode == 1) val = 0.5f * val * (1.f + erff(val * 0.70710678118654752f));
      size_t ci = (size_t)m * N + n;
      if (mode == 2) {
        float wgt  = rowScale[(size_t)m * rsStride];
        float prev = beta ? Cf[ci] : 0.f;
        Cf[ci] = prev + wgt * val;
      } else {
        if (Cf)  Cf[ci]  = val;
        if (Cbf) Cbf[ci] = f2bf(val);
      }
    }
  }
}

// ---------------------------------------------------------------------------
// Flash attention (causal). bf16 Q/K/V in, bf16 ao out.
// One block = one (b,h) x 128 q-rows; 8 waves, 16 q-rows each; KV tiles of 64.
// ---------------------------------------------------------------------------
__global__ __launch_bounds__(256) void flash_attn_kernel(
    const unsigned short* __restrict__ Qg, const unsigned short* __restrict__ Kg,
    const unsigned short* __restrict__ Vg, unsigned short* __restrict__ Og)
{
  __shared__ __align__(16) unsigned short Ql[128][64];   // [q][dh]
  __shared__ __align__(16) unsigned short Kl[64][64];    // [kv][dh]
  __shared__ __align__(16) unsigned short Vl[64][64];    // [dh][kv] (transposed)
  __shared__ __align__(16) unsigned short Pl[8][16][64]; // per-wave P strip

  const int tid  = threadIdx.x;
  const int lane = tid & 31, wave = tid >> 5;
  const int half = lane >> 4, l16 = lane & 15;
  const int bh = blockIdx.y, b = bh >> 4, h = bh & 15;
  const int q0 = blockIdx.x * 128;

  const unsigned short* Qp = Qg + ((size_t)b * T_ + q0) * D_ + h * DH_;
#pragma unroll
  for (int i = 0; i < 4; ++i) {
    int slot = tid + i * 256;                 // 1024 uint4 slots (128x64 bf16)
    int r = slot >> 3, c = (slot & 7) << 3;
    *(uint4*)&Ql[r][c] = *(const uint4*)(Qp + (size_t)r * D_ + c);
  }
  __syncthreads();

  const v8f vzero = {0.f,0.f,0.f,0.f,0.f,0.f,0.f,0.f};
  v8f o[4];
#pragma unroll
  for (int s = 0; s < 4; ++s) o[s] = vzero;
  float mrow[8], lrow[8];
#pragma unroll
  for (int r = 0; r < 8; ++r) { mrow[r] = -1e30f; lrow[r] = 0.f; }

  for (int j0 = 0; j0 < q0 + 128; j0 += 64) {
    const unsigned short* Kp = Kg + ((size_t)b * T_ + j0) * D_ + h * DH_;
    const unsigned short* Vp = Vg + ((size_t)b * T_ + j0) * D_ + h * DH_;
#pragma unroll
    for (int i = 0; i < 2; ++i) {             // K tile: 512 uint4 slots
      int slot = tid + i * 256;
      int r = slot >> 3, c = (slot & 7) << 3;
      *(uint4*)&Kl[r][c] = *(const uint4*)(Kp + (size_t)r * D_ + c);
    }
#pragma unroll
    for (int i = 0; i < 2; ++i) {             // V tile: transpose on store
      int slot = tid + i * 256;
      int r = slot >> 3, c = (slot & 7) << 3;
      uint4 vv = *(const uint4*)(Vp + (size_t)r * D_ + c);
      Vl[c + 0][r] = (unsigned short)(vv.x & 0xFFFFu);
      Vl[c + 1][r] = (unsigned short)(vv.x >> 16);
      Vl[c + 2][r] = (unsigned short)(vv.y & 0xFFFFu);
      Vl[c + 3][r] = (unsigned short)(vv.y >> 16);
      Vl[c + 4][r] = (unsigned short)(vv.z & 0xFFFFu);
      Vl[c + 5][r] = (unsigned short)(vv.z >> 16);
      Vl[c + 6][r] = (unsigned short)(vv.w & 0xFFFFu);
      Vl[c + 7][r] = (unsigned short)(vv.w >> 16);
    }
    __syncthreads();

    // S = Q @ K^T  (16 q-rows x 64 kv-cols per wave)
    v8f sacc[4];
#pragma unroll
    for (int s = 0; s < 4; ++s) sacc[s] = vzero;
#pragma unroll
    for (int ks = 0; ks < 2; ++ks) {
      BF16Frag af;
      const int mr = wave * 16 + l16;
#pragma unroll
      for (int p = 0; p < 8; ++p) {
        int kk = ks * 32 + 2 * p + 8 * half + (p >= 4 ? 8 : 0);
        af.u[p] = *(const unsigned int*)&Ql[mr][kk];
      }
#pragma unroll
      for (int s = 0; s < 4; ++s) {
        BF16Frag bfr;
        const int nr = s * 16 + l16;
#pragma unroll
        for (int p = 0; p < 8; ++p) {
          int kk = ks * 32 + 2 * p + 8 * half + (p >= 4 ? 8 : 0);
          bfr.u[p] = *(const unsigned int*)&Kl[nr][kk];
        }
        sacc[s] = __builtin_amdgcn_wmma_f32_16x16x32_bf16(
            false, af.v, false, bfr.v, (short)0, sacc[s], false, false);
      }
    }

    // scale + causal mask + per-row tile max
    float tmax[8];
#pragma unroll
    for (int r = 0; r < 8; ++r) tmax[r] = -1e30f;
#pragma unroll
    for (int s = 0; s < 4; ++s) {
#pragma unroll
      for (int r = 0; r < 8; ++r) {
        int m = q0 + wave * 16 + r + 8 * half;
        int n = j0 + s * 16 + l16;
        float v = sacc[s][r] * 0.125f;        // 1/sqrt(64)
        if (n > m) v = -1e30f;
        sacc[s][r] = v;
        tmax[r] = fmaxf(tmax[r], v);
      }
    }
#pragma unroll
    for (int r = 0; r < 8; ++r)
      for (int off = 1; off < 16; off <<= 1)
        tmax[r] = fmaxf(tmax[r], __shfl_xor(tmax[r], off, 32));

    float alpha[8], tsum[8];
#pragma unroll
    for (int r = 0; r < 8; ++r) {
      float mn = fmaxf(mrow[r], tmax[r]);
      alpha[r] = __expf(mrow[r] - mn);
      mrow[r] = mn;
      tsum[r] = 0.f;
    }

    // P = exp(S - m) -> bf16 LDS strip
#pragma unroll
    for (int s = 0; s < 4; ++s) {
#pragma unroll
      for (int r = 0; r < 8; ++r) {
        float p = __expf(sacc[s][r] - mrow[r]);
        tsum[r] += p;
        Pl[wave][r + 8 * half][s * 16 + l16] = f2bf(p);
      }
    }
#pragma unroll
    for (int r = 0; r < 8; ++r)
      for (int off = 1; off < 16; off <<= 1)
        tsum[r] += __shfl_xor(tsum[r], off, 32);
#pragma unroll
    for (int r = 0; r < 8; ++r) lrow[r] = lrow[r] * alpha[r] + tsum[r];
#pragma unroll
    for (int s = 0; s < 4; ++s)
#pragma unroll
      for (int r = 0; r < 8; ++r) o[s][r] *= alpha[r];

    __syncthreads();

    // O += P @ V
#pragma unroll
    for (int ks = 0; ks < 2; ++ks) {
      BF16Frag af;
#pragma unroll
      for (int p = 0; p < 8; ++p) {
        int kk = ks * 32 + 2 * p + 8 * half + (p >= 4 ? 8 : 0);
        af.u[p] = *(const unsigned int*)&Pl[wave][l16][kk];
      }
#pragma unroll
      for (int s = 0; s < 4; ++s) {
        BF16Frag bfr;
        const int nr = s * 16 + l16;
#pragma unroll
        for (int p = 0; p < 8; ++p) {
          int kk = ks * 32 + 2 * p + 8 * half + (p >= 4 ? 8 : 0);
          bfr.u[p] = *(const unsigned int*)&Vl[nr][kk];
        }
        o[s] = __builtin_amdgcn_wmma_f32_16x16x32_bf16(
            false, af.v, false, bfr.v, (short)0, o[s], false, false);
      }
    }
    __syncthreads();
  }

#pragma unroll
  for (int s = 0; s < 4; ++s) {
#pragma unroll
    for (int r = 0; r < 8; ++r) {
      int q  = q0 + wave * 16 + r + 8 * half;
      int dh = s * 16 + l16;
      Og[((size_t)b * T_ + q) * D_ + h * DH_ + dh] = f2bf(o[s][r] / lrow[r]);
    }
  }
}

// ---------------------------------------------------------------------------
// out = scale * (a+b) * rsqrt(mean((a+b)^2)+eps); optional bf16 copy
// ---------------------------------------------------------------------------
__global__ __launch_bounds__(256) void rmsnorm_add_kernel(
    const float* __restrict__ a, const float* __restrict__ b,
    const float* __restrict__ scale, float* __restrict__ out,
    unsigned short* __restrict__ outbf)
{
  __shared__ float red[256];
  const int tok = blockIdx.x, tid = threadIdx.x;
  const float* ap = a + (size_t)tok * D_;
  const float* bp = b + (size_t)tok * D_;
  float v[4]; float ss = 0.f;
#pragma unroll
  for (int i = 0; i < 4; ++i) {
    int d = tid + i * 256;
    float x = ap[d] + bp[d];
    v[i] = x; ss += x * x;
  }
  red[tid] = ss; __syncthreads();
  for (int s = 128; s > 0; s >>= 1) {
    if (tid < s) red[tid] += red[tid + s];
    __syncthreads();
  }
  float inv = rsqrtf(red[0] * (1.f / D_) + 1e-6f);
#pragma unroll
  for (int i = 0; i < 4; ++i) {
    int d = tid + i * 256;
    float y = scale[d] * v[i] * inv;
    out[(size_t)tok * D_ + d] = y;
    if (outbf) outbf[(size_t)tok * D_ + d] = f2bf(y);
  }
}

// ---------------------------------------------------------------------------
// router softmax over E=4
// ---------------------------------------------------------------------------
__global__ __launch_bounds__(256) void router_kernel(
    const float* __restrict__ h, const float* __restrict__ wr,
    const float* __restrict__ br, float* __restrict__ wts)
{
  __shared__ float red[4][256];
  const int tok = blockIdx.x, tid = threadIdx.x;
  const float* hp = h + (size_t)tok * D_;
  float p0 = 0, p1 = 0, p2 = 0, p3 = 0;
  for (int d = tid; d < D_; d += 256) {
    float x = hp[d];
    p0 += x * wr[d * 4 + 0]; p1 += x * wr[d * 4 + 1];
    p2 += x * wr[d * 4 + 2]; p3 += x * wr[d * 4 + 3];
  }
  red[0][tid] = p0; red[1][tid] = p1; red[2][tid] = p2; red[3][tid] = p3;
  __syncthreads();
  for (int s = 128; s > 0; s >>= 1) {
    if (tid < s)
      for (int e = 0; e < 4; ++e) red[e][tid] += red[e][tid + s];
    __syncthreads();
  }
  if (tid == 0) {
    float l[4];
    for (int e = 0; e < 4; ++e) l[e] = red[e][0] + br[e];
    float m = fmaxf(fmaxf(l[0], l[1]), fmaxf(l[2], l[3]));
    float ex[4]; float sum = 0.f;
    for (int e = 0; e < 4; ++e) { ex[e] = __expf(l[e] - m); sum += ex[e]; }
    for (int e = 0; e < 4; ++e) wts[(size_t)tok * 4 + e] = ex[e] / sum;
  }
}

// ---------------------------------------------------------------------------
extern "C" void kernel_launch(void* const* d_in, const int* in_sizes, int n_in,
                              void* d_out, int out_size, void* d_ws, size_t ws_size,
                              hipStream_t stream)
{
  (void)in_sizes; (void)n_in; (void)out_size; (void)ws_size;
  const float* x        = (const float*)d_in[0];
  const float* w_kv     = (const float*)d_in[1];
  const float* w_kup    = (const float*)d_in[2];
  const float* w_vup    = (const float*)d_in[3];
  const float* w_q      = (const float*)d_in[4];
  const float* w_o      = (const float*)d_in[5];
  const float* w_router = (const float*)d_in[6];
  const float* b_router = (const float*)d_in[7];
  const float* ew1      = (const float*)d_in[8];
  const float* eb1      = (const float*)d_in[9];
  const float* ew2      = (const float*)d_in[10];
  const float* eb2      = (const float*)d_in[11];
  const float* scale1   = (const float*)d_in[12];
  const float* scale2   = (const float*)d_in[13];

  char* w = (char*)d_ws;
  auto allocB = [&](size_t bytes) -> char* {
    char* p = w; w += (bytes + 255) & ~(size_t)255; return p;
  };
  typedef unsigned short u16;
  u16* xbf    = (u16*)allocB((size_t)NTOK * D_ * 2);
  u16* wkvb   = (u16*)allocB((size_t)D_ * LAT_ * 2);
  u16* wkupb  = (u16*)allocB((size_t)LAT_ * D_ * 2);
  u16* wvupb  = (u16*)allocB((size_t)LAT_ * D_ * 2);
  u16* wqb    = (u16*)allocB((size_t)D_ * D_ * 2);
  u16* wob    = (u16*)allocB((size_t)D_ * D_ * 2);
  u16* ew1b   = (u16*)allocB((size_t)E_ * D_ * EDIM_ * 2);
  u16* ew2b   = (u16*)allocB((size_t)E_ * EDIM_ * D_ * 2);
  u16* latb   = (u16*)allocB((size_t)NTOK * LAT_ * 2);
  u16* Qbf    = (u16*)allocB((size_t)NTOK * D_ * 2);
  u16* Kbf    = (u16*)allocB((size_t)NTOK * D_ * 2);
  u16* Vbf    = (u16*)allocB((size_t)NTOK * D_ * 2);
  u16* aobf   = (u16*)allocB((size_t)NTOK * D_ * 2);
  u16* hbf    = (u16*)allocB((size_t)NTOK * D_ * 2);
  u16* hidbf  = (u16*)allocB((size_t)NTOK * EDIM_ * 2);
  float* aow  = (float*)allocB((size_t)NTOK * D_ * 4);
  float* hbuf = (float*)allocB((size_t)NTOK * D_ * 4);
  float* moe  = (float*)allocB((size_t)NTOK * D_ * 4);
  float* wts  = (float*)allocB((size_t)NTOK * E_ * 4);

  auto cvt = [&](const float* in, u16* out, size_t n) {
    long long n4 = (long long)(n / 4);
    int grid = (int)((n4 + 255) / 256); if (grid > 2048) grid = 2048;
    cvt_f32_bf16<<<grid, 256, 0, stream>>>(in, out, n4);
  };
  cvt(x, xbf, (size_t)NTOK * D_);
  cvt(w_kv, wkvb, (size_t)D_ * LAT_);
  cvt(w_kup, wkupb, (size_t)LAT_ * D_);
  cvt(w_vup, wvupb, (size_t)LAT_ * D_);
  cvt(w_q, wqb, (size_t)D_ * D_);
  cvt(w_o, wob, (size_t)D_ * D_);
  cvt(ew1, ew1b, (size_t)E_ * D_ * EDIM_);
  cvt(ew2, ew2b, (size_t)E_ * EDIM_ * D_);

  auto gemm = [&](const u16* A, const u16* Bm, float* Cf, u16* Cbf,
                  int M, int N, int K, const float* bias, int mode,
                  const float* rs, int rss, int beta) {
    dim3 grid((N + 127) / 128, M / 128);
    gemm_wmma_bf16<<<grid, 256, 0, stream>>>(A, Bm, Cf, Cbf, M, N, K,
                                             bias, mode, rs, rss, beta);
  };

  // MLA projections
  gemm(xbf, wkvb, nullptr, latb, NTOK, LAT_, D_, nullptr, 0, nullptr, 0, 0);
  gemm(latb, wkupb, nullptr, Kbf, NTOK, D_, LAT_, nullptr, 0, nullptr, 0, 0);
  gemm(latb, wvupb, nullptr, Vbf, NTOK, D_, LAT_, nullptr, 0, nullptr, 0, 0);
  gemm(xbf, wqb, nullptr, Qbf, NTOK, D_, D_, nullptr, 0, nullptr, 0, 0);

  // causal flash attention
  flash_attn_kernel<<<dim3(T_ / 128, B_ * H_), 256, 0, stream>>>(Qbf, Kbf, Vbf, aobf);

  // output projection + residual rmsnorm (h kept fp32 + bf16)
  gemm(aobf, wob, aow, nullptr, NTOK, D_, D_, nullptr, 0, nullptr, 0, 0);
  rmsnorm_add_kernel<<<NTOK, 256, 0, stream>>>(x, aow, scale1, hbuf, hbf);

  // router softmax
  router_kernel<<<NTOK, 256, 0, stream>>>(hbuf, w_router, b_router, wts);

  // dense MoE: all 4 experts, weighted accumulate into moe
  for (int e = 0; e < E_; ++e) {
    gemm(hbf, ew1b + (size_t)e * D_ * EDIM_, nullptr, hidbf, NTOK, EDIM_, D_,
         eb1 + (size_t)e * EDIM_, 1, nullptr, 0, 0);
    gemm(hidbf, ew2b + (size_t)e * EDIM_ * D_, moe, nullptr, NTOK, D_, EDIM_,
         eb2 + (size_t)e * D_, 2, wts + e, E_, e > 0);
  }

  // final residual rmsnorm -> d_out (fp32)
  rmsnorm_add_kernel<<<NTOK, 256, 0, stream>>>(hbuf, moe, scale2,
                                               (float*)d_out, nullptr);
}
